// SoftmaxMatcher_27925877358760
// MI455X (gfx1250) — matrix operand order repaired and checked by reading, not compiled, not run on hardware
//
#include <hip/hip_runtime.h>
#include <hip/hip_bf16.h>
#include <math.h>

typedef __attribute__((ext_vector_type(16))) _Float16 v16h;
typedef __attribute__((ext_vector_type(8)))  _Float16 v8h;
typedef __attribute__((ext_vector_type(8)))  float    v8f;

#define NKP   512      // keypoints per frame
#define C_DIM 128      // descriptor channels
#define HW    65536    // 256*256
#define NB    4        // batch pairs

// ---------------------------------------------------------------------------
// Zero the per-row accumulators (denom, sum_u, sum_v): 3 * 2048 floats
// ---------------------------------------------------------------------------
__global__ void zero_acc_kernel(float* __restrict__ acc) {
    int i = blockIdx.x * 256 + threadIdx.x;
    if (i < 3 * NB * NKP) acc[i] = 0.0f;
}

// ---------------------------------------------------------------------------
// Normalize source keypoint descriptors.
//  in : keypoint_desc (8,128,512) f32, frames 0,2,4,6
//  out: src_h  [b][n][c] f16 (L2-normalized)  -- WMMA A-operand friendly
//       src_f  [b][n][c] f32 (L2-normalized)  -- for final dot product
// ---------------------------------------------------------------------------
__global__ void prep_src_kernel(const float* __restrict__ kd,
                                _Float16* __restrict__ src_h,
                                float* __restrict__ src_f) {
    int id = blockIdx.x * 256 + threadIdx.x;   // 0..2047 = b*512+n
    if (id >= NB * NKP) return;
    int b = id >> 9;
    int n = id & (NKP - 1);
    const float* p = kd + (size_t)(2 * b) * C_DIM * NKP + n;  // frame 2b
    float s = 0.0f;
    for (int c = 0; c < C_DIM; ++c) {
        float v = p[c * NKP];
        s += v * v;
    }
    float inv = 1.0f / fmaxf(sqrtf(s), 1e-12f);
    _Float16* q  = src_h + (size_t)id * C_DIM;
    float*    qf = src_f + (size_t)id * C_DIM;
    for (int c = 0; c < C_DIM; ++c) {
        float v = p[c * NKP] * inv;
        q[c]  = (_Float16)v;
        qf[c] = v;
    }
}

// ---------------------------------------------------------------------------
// Target dense descriptors: one coalesced sweep.
//  f32 -> f16 conversion (raw values, layout kept [b][c][m]) +
//  per-pixel inverse L2 norm (normalization is folded into the GEMM epilogue).
// ---------------------------------------------------------------------------
__global__ void prep_tgt_kernel(const float* __restrict__ dd,
                                _Float16* __restrict__ tgt_h,
                                float* __restrict__ inv_t) {
    int b = blockIdx.y;
    int m = blockIdx.x * 256 + threadIdx.x;    // pixel index 0..65535
    const float* p = dd + ((size_t)(2 * b + 1) * C_DIM) * HW + m;  // frame 2b+1
    _Float16*    q = tgt_h + ((size_t)b * C_DIM) * HW + m;
    float s = 0.0f;
    for (int c = 0; c < C_DIM; ++c) {
        float v = p[(size_t)c * HW];
        s += v * v;
        q[(size_t)c * HW] = (_Float16)v;
    }
    inv_t[b * HW + m] = 1.0f / fmaxf(sqrtf(s), 1e-12f);
}

// A-operand loader: 8 contiguous halves + 8 contiguous halves (cdna5 16-bit
// A-matrix 16x32: lane<16 holds K {0..7,16..23}, lane>=16 holds {8..15,24..31})
__device__ __forceinline__ v16h load_a16(const _Float16* p) {
    v8h lo = *(const v8h*)p;
    v8h hi = *(const v8h*)(p + 16);
    return __builtin_shufflevector(lo, hi, 0, 1, 2, 3, 4, 5, 6, 7,
                                           8, 9, 10, 11, 12, 13, 14, 15);
}

// ---------------------------------------------------------------------------
// Two-pass fused WMMA GEMM + temperature softmax.
//   Block = 256 threads = 8 waves. Each wave owns ONE 16-row n-tile; all 8
//   waves share the SAME 1024 m-columns, so B tiles are staged through LDS
//   once per block and reused by all 8 n-tiles (8x B-traffic reduction).
//   K = 128 -> 4x v_wmma_f32_16x16x32_f16 per 16x16 output tile.
//   STORE=false: accumulate softmax stats (denom, sum_u, sum_v) only.
//   STORE=true : recompute, write normalized softmax with NT stores.
// Grid: NB * 4 ngroups * 64 msegs = 1024 blocks.
// ---------------------------------------------------------------------------
template <bool STORE>
__global__ __launch_bounds__(256) void gemm_softmax_kernel(
    const _Float16* __restrict__ src_h,   // [b][n][c] normalized f16
    const _Float16* __restrict__ tgt_h,   // [b][c][m] raw f16
    const float* __restrict__ inv_t,      // [b][m] inverse norms
    float* __restrict__ out_soft,         // [b*512+n][m]
    float* __restrict__ denom,            // [2048]
    float* __restrict__ sumu,             // [2048]
    float* __restrict__ sumv) {           // [2048]
    // staged B: 4 m-tiles (64 cols) x 128 K, f16, tile-major = 16 KB
    __shared__ _Float16 Bs[4 * C_DIM * 16];

    int lin    = blockIdx.x;              // 0..1023
    int b      = lin >> 8;                // 256 blocks per batch
    int rem    = lin & 255;
    int ngroup = rem >> 6;                // 0..3  (128 n-rows each)
    int mseg   = rem & 63;                // 0..63 (1024-column span)
    int wave   = threadIdx.x >> 5;        // 0..7 -> n-tile within n-group
    int lane   = threadIdx.x & 31;
    int lrow   = lane & 15;
    int hhi    = (lane >> 4) & 1;

    int n_base = ngroup * 128 + wave * 16;
    int m_span = mseg * 1024;

    // ---- Load A (this wave's 16 rows x 128 K) into 4 K-chunk reg tiles ----
    const _Float16* ap = src_h + ((size_t)(b * NKP + n_base + lrow) << 7);
    int cofs = hhi * 8;
    v16h a0 = load_a16(ap + 0  + cofs);
    v16h a1 = load_a16(ap + 32 + cofs);
    v16h a2 = load_a16(ap + 64 + cofs);
    v16h a3 = load_a16(ap + 96 + cofs);

    float ps[8], pu[8], pv[8], rden[8];
    int rowb = b * NKP + n_base + hhi * 8;
#pragma unroll
    for (int r = 0; r < 8; ++r) {
        ps[r] = 0.f; pu[r] = 0.f; pv[r] = 0.f;
        if (STORE) rden[r] = 1.0f / denom[rowb + r];
    }

    for (int mg = 0; mg < 16; ++mg) {     // 16 groups of 64 columns
        int mgbase = m_span + mg * 64;
        __syncthreads();                  // previous-group LDS reads done
        // ---- cooperative stage: 1024 16-byte chunks, 4 per thread ----
#pragma unroll
        for (int j = 0; j < 4; ++j) {
            int cid = threadIdx.x + j * 256;   // 0..1023
            int k   = cid >> 3;                // K row 0..127
            int mc  = cid & 7;                 // 8-half chunk within 64 cols
            const v8h* g = (const v8h*)(tgt_h +
                (((size_t)(b * C_DIM + k)) << 16) + mgbase + mc * 8);
            *(v8h*)&Bs[((mc >> 1) * C_DIM + k) * 16 + (mc & 1) * 8] = *g;
        }
        __syncthreads();

#pragma unroll
        for (int mt4 = 0; mt4 < 4; ++mt4) {
            v8f acc = {0.f, 0.f, 0.f, 0.f, 0.f, 0.f, 0.f, 0.f};
            // B operand from LDS: lane%16 = m column, elements gather K rows
            // in table order (8h+s, 16+8h+s-8) within each 32-chunk
            const _Float16* bb = &Bs[(size_t)(mt4 * C_DIM) * 16];
            v16h bm;
            for (int s = 0; s < 16; ++s) bm[s] = bb[(size_t)(0 + hhi * 8 + (s & 7) + (s >> 3) * 16) * 16 + lrow];
            acc = __builtin_amdgcn_wmma_f32_16x16x32_f16(
                false, a0, false, bm, (short)0, acc, false, false);
            for (int s = 0; s < 16; ++s) bm[s] = bb[(size_t)(32 + hhi * 8 + (s & 7) + (s >> 3) * 16) * 16 + lrow];
            acc = __builtin_amdgcn_wmma_f32_16x16x32_f16(
                false, a1, false, bm, (short)0, acc, false, false);
            for (int s = 0; s < 16; ++s) bm[s] = bb[(size_t)(64 + hhi * 8 + (s & 7) + (s >> 3) * 16) * 16 + lrow];
            acc = __builtin_amdgcn_wmma_f32_16x16x32_f16(
                false, a2, false, bm, (short)0, acc, false, false);
            for (int s = 0; s < 16; ++s) bm[s] = bb[(size_t)(96 + hhi * 8 + (s & 7) + (s >> 3) * 16) * 16 + lrow];
            acc = __builtin_amdgcn_wmma_f32_16x16x32_f16(
                false, a3, false, bm, (short)0, acc, false, false);

            // Epilogue: fold target inv-norm, exp((v-1)/T) (cosine sim <= 1
            // so the implicit softmax max is 1: no overflow, no max pass).
            int   m  = mgbase + mt4 * 16 + lrow;  // this lane's column
            float sc = inv_t[b * HW + m];
            float uu = (float)(m & 255);
            float vv = (float)(m >> 8);
            float* op = out_soft + (((size_t)rowb) << 16) + m;
#pragma unroll
            for (int r = 0; r < 8; ++r) {
                float e = __expf((acc[r] * sc - 1.0f) * 100.0f);
                if (STORE) {
                    __builtin_nontemporal_store(e * rden[r],
                                                op + ((size_t)r << 16));
                } else {
                    ps[r] += e;
                    pu[r] += e * uu;
                    pv[r] += e * vv;
                }
            }
        }
    }

    if (!STORE) {
        // Reduce the 16 column-lanes of each half-wave, then a few atomics.
#pragma unroll
        for (int r = 0; r < 8; ++r) {
            float a = ps[r], x = pu[r], y = pv[r];
            for (int msk = 1; msk < 16; msk <<= 1) {
                a += __shfl_xor(a, msk, 32);
                x += __shfl_xor(x, msk, 32);
                y += __shfl_xor(y, msk, 32);
            }
            if (lrow == 0) {
                atomicAdd(&denom[rowb + r], a);
                atomicAdd(&sumu[rowb + r], x);
                atomicAdd(&sumv[rowb + r], y);
            }
        }
    }
}

// ---------------------------------------------------------------------------
// Final per-keypoint epilogue: pseudo coords, bilinear grid-sample of scores
// and 128-channel descriptors, match weights. One 128-thread block per (b,n).
// ---------------------------------------------------------------------------
__device__ __forceinline__ float samp2d(const float* img, float xf, float yf) {
    bool valid = (xf >= 0.f) && (xf <= 255.f) && (yf >= 0.f) && (yf <= 255.f);
    int xi = (int)fminf(fmaxf(xf, 0.f), 255.f);
    int yi = (int)fminf(fmaxf(yf, 0.f), 255.f);
    float v = img[yi * 256 + xi];
    return valid ? v : 0.0f;
}

__global__ __launch_bounds__(128) void finalize_kernel(
    const float* __restrict__ kp_scores,     // (8,1,512)
    const float* __restrict__ scores_dense,  // (8,1,256,256)
    const float* __restrict__ desc_dense,    // (8,128,256,256)
    const float* __restrict__ src_f,         // [b][n][c] normalized f32
    const float* __restrict__ denom,
    const float* __restrict__ sumu,
    const float* __restrict__ sumv,
    float* __restrict__ out_coords,          // (4,512,2)
    float* __restrict__ out_weights) {       // (4,1,512)
    __shared__ float sh[128];
    int id = blockIdx.x;          // b*512+n
    int b  = id >> 9;
    int n  = id & (NKP - 1);
    int c  = threadIdx.x;

    float den = denom[id];
    float puc = sumu[id] / den;
    float pvc = sumv[id] / den;
    if (c == 0) {
        out_coords[2 * id + 0] = puc;
        out_coords[2 * id + 1] = pvc;
    }
    // grid_sample, align_corners=False, zero padding (reference math)
    float un = 2.0f * puc / 255.0f - 1.0f;
    float vn = 2.0f * pvc / 255.0f - 1.0f;
    float x  = ((un + 1.0f) * 256.0f - 1.0f) * 0.5f;
    float y  = ((vn + 1.0f) * 256.0f - 1.0f) * 0.5f;
    float x0 = floorf(x), y0 = floorf(y);
    float wx1 = x - x0, wy1 = y - y0;
    float w00 = (1.f - wx1) * (1.f - wy1);
    float w10 = wx1 * (1.f - wy1);
    float w01 = (1.f - wx1) * wy1;
    float w11 = wx1 * wy1;

    const float* dp = desc_dense + (((size_t)(2 * b + 1) * C_DIM + c) << 16);
    float pd = samp2d(dp, x0, y0) * w00 + samp2d(dp, x0 + 1.f, y0) * w10 +
               samp2d(dp, x0, y0 + 1.f) * w01 +
               samp2d(dp, x0 + 1.f, y0 + 1.f) * w11;

    sh[c] = src_f[(size_t)id * C_DIM + c] * pd;
    __syncthreads();
    for (int s = 64; s > 0; s >>= 1) {
        if (c < s) sh[c] += sh[c + s];
        __syncthreads();
    }
    if (c == 0) {
        const float* sp = scores_dense + ((size_t)(2 * b + 1) << 16);
        float psc = samp2d(sp, x0, y0) * w00 + samp2d(sp, x0 + 1.f, y0) * w10 +
                    samp2d(sp, x0, y0 + 1.f) * w01 +
                    samp2d(sp, x0 + 1.f, y0 + 1.f) * w11;
        float dms = sh[0] / (float)C_DIM;
        out_weights[id] =
            0.5f * (dms + 1.0f) * kp_scores[(size_t)(2 * b) * NKP + n] * psc;
    }
}

// ---------------------------------------------------------------------------
extern "C" void kernel_launch(void* const* d_in, const int* in_sizes, int n_in,
                              void* d_out, int out_size, void* d_ws, size_t ws_size,
                              hipStream_t stream) {
    const float* kp_scores    = (const float*)d_in[0];  // (8,1,512)
    const float* kp_desc      = (const float*)d_in[1];  // (8,128,512)
    const float* scores_dense = (const float*)d_in[2];  // (8,1,256,256)
    const float* desc_dense   = (const float*)d_in[3];  // (8,128,256,256)

    // Workspace layout (bytes):
    //   [0,           512K)  src_h  : 4*512*128 f16
    //   [512K,        1.5M)  src_f  : 4*512*128 f32
    //   [1.5M,        2.5M)  inv_t  : 4*65536  f32
    //   [2.5M,  2.5M+24K)    denom/sumu/sumv : 3*2048 f32
    //   [3M,    3M+64M)      tgt_h  : 4*128*65536 f16
    char* wsb = (char*)d_ws;
    _Float16* src_h = (_Float16*)(wsb);
    float*    src_f = (float*)(wsb + 512 * 1024);
    float*    inv_t = (float*)(wsb + 1536 * 1024);
    float*    accs  = (float*)(wsb + 2560 * 1024);
    float*    denom = accs;
    float*    sumu  = accs + NB * NKP;
    float*    sumv  = accs + 2 * NB * NKP;
    _Float16* tgt_h = (_Float16*)(wsb + 3 * 1024 * 1024);

    // Output layout: coords (4096 f32) | weights (2048 f32) | soft (4*512*65536)
    float* out_coords  = (float*)d_out;
    float* out_weights = out_coords + NB * NKP * 2;
    float* out_soft    = out_weights + NB * NKP;

    zero_acc_kernel<<<24, 256, 0, stream>>>(accs);
    prep_src_kernel<<<(NB * NKP + 255) / 256, 256, 0, stream>>>(kp_desc, src_h,
                                                                src_f);
    {
        dim3 g(HW / 256, NB);
        prep_tgt_kernel<<<g, 256, 0, stream>>>(desc_dense, tgt_h, inv_t);
    }
    // Pass 1: softmax statistics only (no 537MB store).
    gemm_softmax_kernel<false><<<NB * 256, 256, 0, stream>>>(
        src_h, tgt_h, inv_t, out_soft, denom, sumu, sumv);
    // Pass 2: recompute + write normalized softmax once (non-temporal).
    gemm_softmax_kernel<true><<<NB * 256, 256, 0, stream>>>(
        src_h, tgt_h, inv_t, out_soft, denom, sumu, sumv);
    finalize_kernel<<<NB * NKP, 128, 0, stream>>>(kp_scores, scores_dense,
                                                  desc_dense, src_f, denom,
                                                  sumu, sumv, out_coords,
                                                  out_weights);
}
